// SpeciePredictor_8564164788594
// MI455X (gfx1250) — matrix operand order repaired and checked by reading, not compiled, not run on hardware
//
#include <hip/hip_runtime.h>
#include <hip/hip_bf16.h>

// ---------------------------------------------------------------------------
// MI455X (gfx1250, wave32) fully-fused forward pass.
// One block = 16 batch rows; all intermediates live in LDS; every GEMM stage
// runs on v_wmma_f32_16x16x32_bf16. Weights are pre-packed to bf16 WMMA
// B-fragment order in d_ws by pack_w().
// ---------------------------------------------------------------------------

typedef __bf16 bf16_t;
typedef __attribute__((ext_vector_type(16))) __bf16 v16bf;
typedef __attribute__((ext_vector_type(8)))  float  v8f;

#define WAVES 8   // 256 threads / wave32

// ---- packed-weight offsets in d_ws (units: bf16 elements) ------------------
static constexpr size_t OFS_DNA = 0;        // 768x256 : kt24 nt16
static constexpr size_t OFS_IMG = 196608;   // 512x256 : kt16 nt16
static constexpr size_t OFS_QKV = 327680;   // 256x768 : kt8  nt48
static constexpr size_t OFS_O   = 524288;   // 256x256 : kt8  nt16
static constexpr size_t OFS_F1  = 589824;   // 256x256
static constexpr size_t OFS_F2  = 655360;   // 256x256
static constexpr size_t OFS_D1  = 720896;   // 256x744 : kt8  nt47
static constexpr size_t OFS_D2  = 913408;   // 744x372 : kt24 nt24 (K zero-padded)
static constexpr size_t OFS_FP  = 1208320;  // 256x128 : kt8  nt8
static constexpr size_t OFS_GE  = 1241088;  // 372x64  : kt12 nt4  (K zero-padded)
static constexpr size_t OFS_S1  = 1265664;  // 192x64  : kt6  nt4
static constexpr size_t OFS_S2  = 1277952;  // 64x23   : kt2  nt2  (N zero-padded)
// total 1,280,000 bf16 = 2.56 MB

// ---------------------------------------------------------------------------
// Repack fp32 row-major W[K][N] into WMMA bf16 B fragments:
//   frag = nt*ktiles + kt ; within a fragment lane L owns 16 contiguous halves:
//   lanes 0-15 : N = nt*16+L,    K = kt*32 + (0..15)
//   lanes 16-31: N = nt*16+L-16, K = kt*32 + (16..31)
// Out-of-range K/N -> 0 (ragged-dim padding).
// ---------------------------------------------------------------------------
__global__ void pack_w(const float* __restrict__ W, bf16_t* __restrict__ out,
                       int K, int N, int ktiles, int ntiles) {
    int idx = blockIdx.x * blockDim.x + threadIdx.x;
    int total = ktiles * ntiles * 512;
    if (idx >= total) return;
    int j    = idx & 15;
    int lane = (idx >> 4) & 31;
    int frag = idx >> 9;
    int kt = frag % ktiles;
    int nt = frag / ktiles;
    int n = nt * 16 + (lane & 15);
    int k = kt * 32 + ((lane < 16) ? j : 16 + j);
    float v = (k < K && n < N) ? W[(size_t)k * N + n] : 0.0f;
    out[idx] = (bf16_t)v;
}

// ---------------------------------------------------------------------------
// 16xN GEMM stage: C = act(A[16|32 x K]@Wp + bias), A bf16 in LDS.
// Epilogue fans out to (optional) LDS-bf16, LDS-f32 and global-f32 targets.
// ---------------------------------------------------------------------------
__device__ __forceinline__ void gemm16(
    const bf16_t* __restrict__ ldsA, int sA,           // A row stride (halves)
    const bf16_t* __restrict__ wpB, int ktiles, int ntiles, int mtiles,
    const float* __restrict__ bias, int nvalid, int act,  // 0 none,1 relu,2 sigmoid
    bf16_t* cBf, int sBf, int colOfs,
    float* cF, int sF,
    float* gOut, int gStride,
    int wave, int lane)
{
    const int ntot = mtiles * ntiles;
    for (int t = wave; t < ntot; t += WAVES) {
        const int mt = t / ntiles;
        const int nt = t - mt * ntiles;
        v8f acc = {0.f, 0.f, 0.f, 0.f, 0.f, 0.f, 0.f, 0.f};
        // A fragment: lane<16 -> K 0-15 ; lane>=16 -> K 16-31
        const bf16_t* arow = ldsA + (size_t)(mt * 16 + (lane & 15)) * sA
                                  + ((lane < 16) ? 0 : 16);
        const bf16_t* bptr = wpB + ((size_t)nt * ktiles * 32 + lane) * 16;
        for (int kt = 0; kt < ktiles; ++kt) {
            union { v16bf v; uint4 u[2]; } afrag, bfrag;
            afrag.u[0] = *(const uint4*)(arow + kt * 32);
            afrag.u[1] = *(const uint4*)(arow + kt * 32 + 8);
            bfrag.u[0] = ((const uint4*)bptr)[0];
            bfrag.u[1] = ((const uint4*)bptr)[1];
            bptr += 512;
            acc = __builtin_amdgcn_wmma_f32_16x16x32_bf16(
                false, afrag.v, false, bfrag.v, (short)0, acc, false, false);
        }
        // C/D layout: lane&15 = N ; lane>>4 selects M 0-7 vs 8-15 ; vgpr r = M&7
        const int n  = nt * 16 + (lane & 15);
        const int mb = mt * 16 + ((lane >> 4) << 3);
        const bool nv = (n < nvalid);
        const float bv = nv ? bias[n] : 0.0f;
#pragma unroll
        for (int r = 0; r < 8; ++r) {
            float v = acc[r] + bv;
            if (act == 1) v = fmaxf(v, 0.0f);
            if (act == 2) v = 1.0f / (1.0f + __expf(-v));
            if (!nv) v = 0.0f;                 // zero-fill ragged-N padding
            const int m = mb + r;
            if (cBf)          cBf[(size_t)m * sBf + colOfs + n] = (bf16_t)v;
            if (cF)           cF[(size_t)m * sF + n] = v;
            if (gOut && nv)   gOut[(size_t)m * gStride + n] = v;
        }
    }
}

// ---------------------------------------------------------------------------
// Megakernel: 16 rows per block, everything in LDS (119 KB dynamic).
// ---------------------------------------------------------------------------
__global__ __launch_bounds__(256)
void bioclip_fused_fwd(
    const float* __restrict__ dna,  const float* __restrict__ img,
    const float* __restrict__ b_dna, const float* __restrict__ b_img,
    const float* __restrict__ b_qkv, const float* __restrict__ b_o,
    const float* __restrict__ b_f1,  const float* __restrict__ b_f2,
    const float* __restrict__ b_d1,  const float* __restrict__ b_d2,
    const float* __restrict__ b_fp,  const float* __restrict__ b_ge,
    const float* __restrict__ b_s1,  const float* __restrict__ b_s2,
    const bf16_t* __restrict__ wp,
    float* __restrict__ oSpec, float* __restrict__ oLoc,
    float* __restrict__ oGen,  float* __restrict__ oFus,
    float* __restrict__ oRed)
{
    extern __shared__ char smem[];
    const int tid  = threadIdx.x;
    const int lane = tid & 31;
    const int wave = tid >> 5;
    const size_t row0 = (size_t)blockIdx.x * 16;

    // persistent region
    float*  sGlogF = (float*)(smem + 0);          // [16][384] f32
    bf16_t* sGlogB = (bf16_t*)(smem + 24576);     // [16][392] bf16
    bf16_t* sFused = (bf16_t*)(smem + 37120);     // [16][264]
    bf16_t* sCat   = (bf16_t*)(smem + 45568);     // [16][200] (gemb|reduced)
    float*  sLocal = (float*)(smem + 51968);      // [16][32]
    float*  sScore = (float*)(smem + 54016);      // [64][4]
    // transient region (aliased across stages)
    char* T = smem + 55040;
    bf16_t* sSeq = (bf16_t*)(T + 0);              // [32][264]
    bf16_t* sCtx = (bf16_t*)(T + 0);              // [16][264] (after seq dead)
    bf16_t* sS1h = (bf16_t*)(T + 0);              // [16][72]  (late)
    bf16_t* sQkv = (bf16_t*)(T + 16896);          // [32][776]
    bf16_t* sDna = (bf16_t*)(T + 16896);          // [16][776] (before qkv)
    bf16_t* sImg = (bf16_t*)(T + 41728);          // [16][520]
    bf16_t* sF0  = (bf16_t*)(T + 16896);          // [16][264] (after qkv dead)
    bf16_t* sH1  = (bf16_t*)(T + 25344);          // [16][264]
    bf16_t* sHd  = (bf16_t*)(T + 33792);          // [16][776]

    // ---- stage A: stage inputs fp32->bf16 into LDS -------------------------
    for (int i = tid; i < 16 * 768; i += 256) {
        int r = i / 768, c = i - r * 768;
        sDna[r * 776 + c] = (bf16_t)dna[(row0 + r) * 768 + c];
    }
    for (int i = tid; i < 16 * 512; i += 256) {
        int r = i >> 9, c = i & 511;
        sImg[r * 520 + c] = (bf16_t)img[(row0 + r) * 512 + c];
    }
    __syncthreads();

    // ---- projections -> seq [token0 rows 0-15 | token1 rows 16-31] ---------
    gemm16(sDna, 776, wp + OFS_DNA, 24, 16, 1, b_dna, 256, 0,
           sSeq, 264, 0, nullptr, 0, nullptr, 0, wave, lane);
    gemm16(sImg, 520, wp + OFS_IMG, 16, 16, 1, b_img, 256, 0,
           sSeq + 16 * 264, 264, 0, nullptr, 0, nullptr, 0, wave, lane);
    __syncthreads();

    // ---- qkv (M = 32 : both tokens) ----------------------------------------
    gemm16(sSeq, 264, wp + OFS_QKV, 8, 48, 2, b_qkv, 768, 0,
           sQkv, 776, 0, nullptr, 0, nullptr, 0, wave, lane);
    __syncthreads();

    // ---- attention (seq len 2, 4 heads, dim 64) ----------------------------
    {   // raw scores: 64 (b,h) pairs x 4 (i,j)
        int pair = tid >> 2, sub = tid & 3;
        int b = pair >> 2, h = pair & 3;
        int i = sub >> 1, j = sub & 1;
        const bf16_t* q = sQkv + (size_t)(i * 16 + b) * 776 + h * 64;
        const bf16_t* k = sQkv + (size_t)(j * 16 + b) * 776 + 256 + h * 64;
        float s = 0.f;
        for (int d = 0; d < 64; ++d) s += (float)q[d] * (float)k[d];
        sScore[pair * 4 + sub] = s * 0.125f;   // 1/sqrt(64)
    }
    __syncthreads();
    if (tid < 64) {  // softmax rows + fold mean-over-queries (linear) into v-weights
        float s00 = sScore[tid*4+0], s01 = sScore[tid*4+1];
        float s10 = sScore[tid*4+2], s11 = sScore[tid*4+3];
        float m0 = fmaxf(s00, s01), m1 = fmaxf(s10, s11);
        float e00 = __expf(s00-m0), e01 = __expf(s01-m0);
        float e10 = __expf(s10-m1), e11 = __expf(s11-m1);
        float a00 = e00/(e00+e01), a01 = e01/(e00+e01);
        float a10 = e10/(e10+e11), a11 = e11/(e10+e11);
        sScore[tid*4+0] = 0.5f*(a00+a10);      // weight on v(token0)
        sScore[tid*4+1] = 0.5f*(a01+a11);      // weight on v(token1)
    }
    __syncthreads();
    for (int i = tid; i < 16 * 256; i += 256) {  // averaged context [16][256]
        int b = i >> 8, c = i & 255, h = c >> 6;
        float w0 = sScore[(b*4+h)*4 + 0], w1 = sScore[(b*4+h)*4 + 1];
        float v0 = (float)sQkv[(size_t)b * 776 + 512 + c];
        float v1 = (float)sQkv[(size_t)(16+b) * 776 + 512 + c];
        sCtx[b * 264 + c] = (bf16_t)(w0 * v0 + w1 * v1);
    }
    __syncthreads();

    // ---- output proj + MLP -------------------------------------------------
    gemm16(sCtx, 264, wp + OFS_O, 8, 16, 1, b_o, 256, 0,
           sF0, 264, 0, nullptr, 0, nullptr, 0, wave, lane);
    __syncthreads();
    gemm16(sF0, 264, wp + OFS_F1, 8, 16, 1, b_f1, 256, 1 /*relu*/,
           sH1, 264, 0, nullptr, 0, nullptr, 0, wave, lane);
    __syncthreads();
    gemm16(sH1, 264, wp + OFS_F2, 8, 16, 1, b_f2, 256, 0,
           sFused, 264, 0, nullptr, 0, oFus + row0 * 256, 256, wave, lane);
    __syncthreads();

    // ---- genus decoder -----------------------------------------------------
    gemm16(sFused, 264, wp + OFS_D1, 8, 47, 1, b_d1, 744, 2 /*sigmoid*/,
           sHd, 776, 0, nullptr, 0, nullptr, 0, wave, lane);
    for (int i = tid; i < 16 * 16; i += 256)        // zero K-pad cols 752..767
        sHd[(size_t)(i >> 4) * 776 + 752 + (i & 15)] = (bf16_t)0.f;
    __syncthreads();
    gemm16(sHd, 776, wp + OFS_D2, 24, 24, 1, b_d2, 372, 0,
           nullptr, 0, 0, sGlogF, 384, nullptr, 0, wave, lane);
    __syncthreads();
    {   // softmax over 372, 16 lanes per row (half-wave shuffles)
        int r = tid >> 4, l = tid & 15;
        float mx = -3.0e38f;
        for (int c = l; c < 372; c += 16) mx = fmaxf(mx, sGlogF[r*384+c]);
        for (int o = 8; o; o >>= 1) mx = fmaxf(mx, __shfl_xor(mx, o, 16));
        float sum = 0.f;
        for (int c = l; c < 372; c += 16) {
            float e = __expf(sGlogF[r*384+c] - mx);
            sGlogF[r*384+c] = e; sum += e;
        }
        for (int o = 8; o; o >>= 1) sum += __shfl_xor(sum, o, 16);
        float inv = 1.0f / sum;
        for (int c = l; c < 384; c += 16) {
            float p = (c < 372) ? sGlogF[r*384+c] * inv : 0.0f;
            sGlogF[r*384+c] = p;
            sGlogB[r*392+c] = (bf16_t)p;
            if (c < 372) oGen[(row0 + r) * 372 + c] = p;
        }
    }
    __syncthreads();

    // ---- species head ------------------------------------------------------
    gemm16(sFused, 264, wp + OFS_FP, 8, 8, 1, b_fp, 128, 0,
           sCat, 200, 64, nullptr, 0, oRed + row0 * 128, 128, wave, lane);
    gemm16(sGlogB, 392, wp + OFS_GE, 12, 4, 1, b_ge, 64, 0,
           sCat, 200, 0, nullptr, 0, nullptr, 0, wave, lane);
    __syncthreads();
    gemm16(sCat, 200, wp + OFS_S1, 6, 4, 1, b_s1, 64, 2 /*sigmoid*/,
           sS1h, 72, 0, nullptr, 0, nullptr, 0, wave, lane);
    __syncthreads();
    gemm16(sS1h, 72, wp + OFS_S2, 2, 2, 1, b_s2, 23, 0,
           nullptr, 0, 0, sLocal, 32, oLoc + row0 * 23, 23, wave, lane);
    __syncthreads();

    // ---- flatten: species[s] = genus[s%372] * local[s/372] -----------------
    for (int i = tid; i < 16 * 1050; i += 256) {
        int r = i / 1050, s = i - r * 1050;
        int g = s % 372, l = s / 372;
        oSpec[(row0 + r) * 1050 + s] = sGlogF[r * 384 + g] * sLocal[r * 32 + l];
    }
}

// ---------------------------------------------------------------------------
extern "C" void kernel_launch(void* const* d_in, const int* in_sizes, int n_in,
                              void* d_out, int out_size, void* d_ws, size_t ws_size,
                              hipStream_t stream) {
    (void)n_in; (void)out_size; (void)ws_size;
    const size_t B = (size_t)in_sizes[0] / 768;
    bf16_t* wp = (bf16_t*)d_ws;

    // --- repack all weights to bf16 WMMA fragment order (L2-resident, 2.56MB)
    struct PW { int idx; size_t ofs; int K, N, kt, nt; };
    const PW tab[12] = {
        { 2, OFS_DNA, 768, 256, 24, 16}, { 4, OFS_IMG, 512, 256, 16, 16},
        { 6, OFS_QKV, 256, 768,  8, 48}, { 8, OFS_O,   256, 256,  8, 16},
        {10, OFS_F1,  256, 256,  8, 16}, {12, OFS_F2,  256, 256,  8, 16},
        {14, OFS_D1,  256, 744,  8, 47}, {16, OFS_D2,  744, 372, 24, 24},
        {18, OFS_FP,  256, 128,  8,  8}, {20, OFS_GE,  372,  64, 12,  4},
        {22, OFS_S1,  192,  64,  6,  4}, {24, OFS_S2,   64,  23,  2,  2}};
    for (int i = 0; i < 12; ++i) {
        int total = tab[i].kt * tab[i].nt * 512;
        pack_w<<<(total + 255) / 256, 256, 0, stream>>>(
            (const float*)d_in[tab[i].idx], wp + tab[i].ofs,
            tab[i].K, tab[i].N, tab[i].kt, tab[i].nt);
    }

    // --- output slices (flat concat in reference return order, fp32)
    float* out   = (float*)d_out;
    float* oSpec = out;
    float* oLoc  = out + B * 1050;
    float* oGen  = out + B * 1073;
    float* oFus  = out + B * 1445;
    float* oRed  = out + B * 1701;

    const int smemBytes = 121600;   // 119 KB of the 320 KB/WGP LDS
    hipFuncSetAttribute(reinterpret_cast<const void*>(bioclip_fused_fwd),
                        hipFuncAttributeMaxDynamicSharedMemorySize, smemBytes);
    bioclip_fused_fwd<<<(unsigned)(B / 16), 256, smemBytes, stream>>>(
        (const float*)d_in[0],  (const float*)d_in[1],
        (const float*)d_in[3],  (const float*)d_in[5],
        (const float*)d_in[7],  (const float*)d_in[9],
        (const float*)d_in[11], (const float*)d_in[13],
        (const float*)d_in[15], (const float*)d_in[17],
        (const float*)d_in[19], (const float*)d_in[21],
        (const float*)d_in[23], (const float*)d_in[25],
        wp, oSpec, oLoc, oGen, oFus, oRed);
}
